// RvT_56916906607127
// MI455X (gfx1250) — compile-verified, matrix operand order checked by reading, not run on hardware
//
#include <hip/hip_runtime.h>
#include <hip/hip_bf16.h>
#include <math.h>

typedef __attribute__((ext_vector_type(16))) _Float16 v16h;
typedef __attribute__((ext_vector_type(8)))  float    v8f;

#define LDK   40          // padded LDS k-stride (halves); row pitch 80B
#define NB    16
#define CIN   128
#define CC    256
#define HH    36
#define WW    36
#define NPIX  1296        // 36*36
#define DHV   34
#define ND    1156        // 34*34
#define DPAD  1280        // 10*128 padded attention depth (multiple of 32)
#define PPAD  1408        // 11*128 padded pixel count
#define TWOC  512
#define KCONV 4608        // 512*9
#define KGATE 4864        // 4608+256
#define TM    128         // output tile M
#define TN    128         // output tile N

// ---------------- WMMA helpers ----------------

__device__ inline v16h frag_ld(const _Float16* base, int row0, int lane) {
  // 16-bit fragment layout: lanes 0-15 -> k 0..7,16..23 ; lanes 16-31 -> k 8..15,24..31
  const _Float16* p = base + (size_t)(row0 + (lane & 15)) * LDK + ((lane & 16) ? 8 : 0);
  v16h f;
#pragma unroll
  for (int j = 0; j < 8; ++j) { f[j] = p[j]; f[j + 8] = p[16 + j]; }
  return f;
}

// wave computes 32(M) x 64(N): 2 A-frags x 4 B-frags -> 8 WMMA per K-step
__device__ inline void wmma_tile8(const _Float16* As, const _Float16* Bs,
                                  int wm, int wn, int lane, v8f acc[2][4]) {
  v16h a[2], b[4];
#pragma unroll
  for (int i = 0; i < 2; ++i) a[i] = frag_ld(As, wm * 32 + i * 16, lane);
#pragma unroll
  for (int j = 0; j < 4; ++j) b[j] = frag_ld(Bs, wn * 64 + j * 16, lane);
#pragma unroll
  for (int i = 0; i < 2; ++i)
#pragma unroll
    for (int j = 0; j < 4; ++j)
      acc[i][j] = __builtin_amdgcn_wmma_f32_16x16x32_f16(
          false, a[i], false, b[j], (short)0, acc[i][j], false, false);
}

__device__ inline void acc_zero(v8f acc[2][4]) {
#pragma unroll
  for (int i = 0; i < 2; ++i)
#pragma unroll
    for (int j = 0; j < 4; ++j)
#pragma unroll
      for (int e = 0; e < 8; ++e) acc[i][j][e] = 0.f;
}

// 32B vector load / LDS store (16 halves = one staged chunk per thread)
__device__ inline void g_ld32(const _Float16* s, uint4& d0, uint4& d1) {
  d0 = ((const uint4*)s)[0];
  d1 = ((const uint4*)s)[1];
}
__device__ inline void lds_st32(_Float16* lds, int r, int ko, uint4 d0, uint4 d1) {
  *(uint4*)&lds[r * LDK + ko]     = d0;
  *(uint4*)&lds[r * LDK + ko + 8] = d1;
}
__device__ inline void zero32(uint4& d0, uint4& d1) {
  d0.x = d0.y = d0.z = d0.w = 0u; d1 = d0;
}

// epilogue: 8 consecutive f16 (one 16B store)
__device__ inline void st8h(_Float16* dst, v8f a) {
  _Float16 h[8];
#pragma unroll
  for (int r = 0; r < 8; ++r) h[r] = (_Float16)a[r];
  *(uint4*)dst = *(const uint4*)h;
}

// ---------------- Prep: f32 NCHW slab -> f16 pixel-major (32x32 LDS transpose) ----------------
// dst[n][p][colofs + c], p in [0,PPAD); p >= NPIX rows get zeros.

__global__ __launch_bounds__(256) void k_tr(const float* __restrict__ src,
                                            _Float16* __restrict__ dst,
                                            int C, int dststride, int colofs) {
  __shared__ float tile[32][33];
  const int n = blockIdx.z;
  const int p0 = blockIdx.x * 32, c0 = blockIdx.y * 32;
  const int tx = threadIdx.x, ty = threadIdx.y;    // (32, 8)
#pragma unroll
  for (int i = 0; i < 4; ++i) {
    const int p = p0 + tx, c = c0 + ty + i * 8;
    tile[ty + i * 8][tx] = (p < NPIX) ? src[((size_t)n * C + c) * NPIX + p] : 0.f;
  }
  __syncthreads();
#pragma unroll
  for (int i = 0; i < 4; ++i) {
    const int p = p0 + ty + i * 8, c = c0 + tx;
    dst[((size_t)n * PPAD + p) * dststride + colofs + c] = (_Float16)tile[tx][ty + i * 8];
  }
}

// ---------------- Prep: weight packing ----------------

__global__ void k_pack_lin(const float* __restrict__ W, _Float16* __restrict__ out, int total) {
  int i = blockIdx.x * 256 + threadIdx.x;
  if (i < total) out[i] = (_Float16)W[i];
}

// conv weight [256][512][3][3] -> [256][kstride] with k = tap*512+ci; optional Wa append at 4608
__global__ void k_pack_conv(const float* __restrict__ W, const float* __restrict__ Wa,
                            _Float16* __restrict__ out, int kstride) {
  int i = blockIdx.x * 256 + threadIdx.x;
  const int total = CC * kstride;
  if (i >= total) return;
  const int m = i / kstride, k = i % kstride;
  float v;
  if (k < KCONV) {
    const int tap = k / TWOC, ci = k % TWOC;
    v = W[((size_t)m * TWOC + ci) * 9 + tap];
  } else {
    v = Wa[(size_t)m * CC + (k - KCONV)];
  }
  out[i] = (_Float16)v;
}

__global__ void k_fill16(uint4* __restrict__ p, size_t n16) {
  size_t i = (size_t)blockIdx.x * 256 + threadIdx.x;
  if (i < n16) { uint4 z; z.x = z.y = z.z = z.w = 0u; p[i] = z; }
}

// ---------------- GEMM 1: xh_t[:, :, 0:256] = Wx * x_in  (K=128) ----------------

__global__ __launch_bounds__(256) void k_proj(const _Float16* __restrict__ Wxp,
                                              const _Float16* __restrict__ xin_t,
                                              _Float16* __restrict__ xh_t) {
  __shared__ _Float16 As[TM * LDK];
  __shared__ _Float16 Bs[TN * LDK];
  const int tid = threadIdx.x, lane = tid & 31, wave = tid >> 5;
  const int wm = wave >> 1, wn = wave & 1;
  const int mt = blockIdx.x, pt = blockIdx.y, n = blockIdx.z;
  const int r = tid >> 1, ko = (tid & 1) * 16;
  const _Float16* Arow = Wxp + (size_t)(mt * TM + r) * CIN;
  const _Float16* Brow = xin_t + ((size_t)n * PPAD + pt * TN + r) * CIN;
  v8f acc[2][4]; acc_zero(acc);

  uint4 a0, a1, b0, b1;
  g_ld32(Arow + ko, a0, a1);
  g_ld32(Brow + ko, b0, b1);
  for (int kb = 0; kb < CIN / 32; ++kb) {
    __syncthreads();
    lds_st32(As, r, ko, a0, a1);
    lds_st32(Bs, r, ko, b0, b1);
    __syncthreads();
    if (kb + 1 < CIN / 32) {
      g_ld32(Arow + (kb + 1) * 32 + ko, a0, a1);
      g_ld32(Brow + (kb + 1) * 32 + ko, b0, b1);
    }
    wmma_tile8(As, Bs, wm, wn, lane, acc);
  }
  const int mb = mt * TM + wm * 32, nb = pt * TN + wn * 64;
#pragma unroll
  for (int i = 0; i < 2; ++i)
#pragma unroll
    for (int j = 0; j < 4; ++j) {
      const int m = mb + i * 16 + ((lane & 16) ? 8 : 0);
      const int p = nb + j * 16 + (lane & 15);
      st8h(xh_t + ((size_t)n * PPAD + p) * TWOC + m, acc[i][j]);  // pad rows get zeros
    }
}

// ---------------- GEMM 2: conv3x3 implicit-im2col (K=4608) ----------------
// transposed=1: out_t[n][p][m] (rows_out rows). transposed=0: out[n][m][DPAD] (for v).

__global__ __launch_bounds__(256) void k_conv3(const _Float16* __restrict__ Wp,
                                               const _Float16* __restrict__ xh_t,
                                               _Float16* __restrict__ out,
                                               int outW, int outN, int pad,
                                               int transposed, int rows_out) {
  __shared__ _Float16 As[TM * LDK];
  __shared__ _Float16 Bs[TN * LDK];
  const int tid = threadIdx.x, lane = tid & 31, wave = tid >> 5;
  const int wm = wave >> 1, wn = wave & 1;
  const int mt = blockIdx.x, pt = blockIdx.y, n = blockIdx.z;
  const int r = tid >> 1, ko = (tid & 1) * 16;
  const int p = pt * TN + r;
  const int y = p / outW, x = p % outW;
  const bool pv = (p < outN);
  const _Float16* Arow = Wp + (size_t)(mt * TM + r) * KCONV;
  const _Float16* Xb = xh_t + (size_t)n * PPAD * TWOC;
  v8f acc[2][4]; acc_zero(acc);

  uint4 a0, a1, b0, b1;
  auto loadB = [&](int kb, uint4& d0, uint4& d1) {
    const int tap = kb >> 4;
    const int iy = y + tap / 3 - pad, ix = x + tap % 3 - pad;
    const bool ok = pv && iy >= 0 && iy < HH && ix >= 0 && ix < WW;
    const int prow = ok ? iy * WW + ix : 0;
    g_ld32(Xb + (size_t)prow * TWOC + (kb & 15) * 32 + ko, d0, d1);
    if (!ok) zero32(d0, d1);
  };
  g_ld32(Arow + ko, a0, a1);
  loadB(0, b0, b1);
  for (int kb = 0; kb < 144; ++kb) {
    __syncthreads();
    lds_st32(As, r, ko, a0, a1);
    lds_st32(Bs, r, ko, b0, b1);
    __syncthreads();
    if (kb + 1 < 144) {
      g_ld32(Arow + (kb + 1) * 32 + ko, a0, a1);
      loadB(kb + 1, b0, b1);
    }
    wmma_tile8(As, Bs, wm, wn, lane, acc);
  }
  const int mb = mt * TM + wm * 32, nb = pt * TN + wn * 64;
  if (transposed) {
#pragma unroll
    for (int i = 0; i < 2; ++i)
#pragma unroll
      for (int j = 0; j < 4; ++j) {
        const int m = mb + i * 16 + ((lane & 16) ? 8 : 0);
        const int pr = nb + j * 16 + (lane & 15);
        st8h(out + ((size_t)n * rows_out + pr) * CC + m, acc[i][j]);  // pad rows -> zeros
      }
  } else {
#pragma unroll
    for (int i = 0; i < 2; ++i)
#pragma unroll
      for (int j = 0; j < 4; ++j)
#pragma unroll
        for (int rr = 0; rr < 8; ++rr) {
          const int m = mb + i * 16 + rr + ((lane & 16) ? 8 : 0);
          const int pc = nb + j * 16 + (lane & 15);
          out[((size_t)n * CC + m) * DPAD + pc] = (_Float16)acc[i][j][rr];  // pad cols -> zeros
        }
  }
}

// ---------------- GEMM 3: scores = tau * q^T k  (K=256) ----------------

__global__ __launch_bounds__(256) void k_scores(const _Float16* __restrict__ q_t,
                                                const _Float16* __restrict__ k_t,
                                                const float* __restrict__ tau,
                                                float* __restrict__ scores) {
  __shared__ _Float16 As[TM * LDK];
  __shared__ _Float16 Bs[TN * LDK];
  const int tid = threadIdx.x, lane = tid & 31, wave = tid >> 5;
  const int wm = wave >> 1, wn = wave & 1;
  const int qt = blockIdx.x, dt = blockIdx.y, n = blockIdx.z;
  const int r = tid >> 1, ko = (tid & 1) * 16;
  const float t = tau[0];
  const _Float16* Arow = q_t + ((size_t)n * PPAD + qt * TM + r) * CC;
  const _Float16* Brow = k_t + ((size_t)n * DPAD + dt * TN + r) * CC;
  v8f acc[2][4]; acc_zero(acc);

  uint4 a0, a1, b0, b1;
  g_ld32(Arow + ko, a0, a1);
  g_ld32(Brow + ko, b0, b1);
  for (int kb = 0; kb < CC / 32; ++kb) {
    __syncthreads();
    lds_st32(As, r, ko, a0, a1);
    lds_st32(Bs, r, ko, b0, b1);
    __syncthreads();
    if (kb + 1 < CC / 32) {
      g_ld32(Arow + (kb + 1) * 32 + ko, a0, a1);
      g_ld32(Brow + (kb + 1) * 32 + ko, b0, b1);
    }
    wmma_tile8(As, Bs, wm, wn, lane, acc);
  }
  const int mb = qt * TM + wm * 32, nb = dt * TN + wn * 64;
#pragma unroll
  for (int i = 0; i < 2; ++i)
#pragma unroll
    for (int j = 0; j < 4; ++j)
#pragma unroll
      for (int rr = 0; rr < 8; ++rr) {
        const int qp = mb + i * 16 + rr + ((lane & 16) ? 8 : 0);
        const int d  = nb + j * 16 + (lane & 15);
        if (qp < NPIX) scores[((size_t)n * NPIX + qp) * DPAD + d] = t * acc[i][j][rr];
      }
}

// ---------------- softmax over D=1156 rows; probs padded [n][PPAD][DPAD] ----------------

__global__ __launch_bounds__(256) void k_softmax(const float* __restrict__ scores,
                                                 _Float16* __restrict__ probs) {
  const int row = blockIdx.x;                 // 0..NB*NPIX-1
  const int n = row / NPIX, qp = row % NPIX;
  const float* s = scores + ((size_t)n * NPIX + qp) * DPAD;
  _Float16* p = probs + ((size_t)n * PPAD + qp) * DPAD;
  __shared__ float red[256];
  const int tid = threadIdx.x;
  float mx = -3.0e38f;
  for (int d = tid; d < ND; d += 256) mx = fmaxf(mx, s[d]);
  red[tid] = mx; __syncthreads();
  for (int off = 128; off > 0; off >>= 1) {
    if (tid < off) red[tid] = fmaxf(red[tid], red[tid + off]);
    __syncthreads();
  }
  mx = red[0]; __syncthreads();
  float sum = 0.f;
  for (int d = tid; d < ND; d += 256) sum += __expf(s[d] - mx);
  red[tid] = sum; __syncthreads();
  for (int off = 128; off > 0; off >>= 1) {
    if (tid < off) red[tid] += red[tid + off];
    __syncthreads();
  }
  const float inv = 1.f / red[0];
  for (int d = tid; d < DPAD; d += 256)
    p[d] = (_Float16)((d < ND) ? __expf(s[d] - mx) * inv : 0.f);
}

// ---------------- GEMM 4: a_t = (v @ w^T)^T  (K=DPAD) ----------------

__global__ __launch_bounds__(256) void k_apply(const _Float16* __restrict__ v,
                                               const _Float16* __restrict__ probs,
                                               _Float16* __restrict__ a_t) {
  __shared__ _Float16 As[TM * LDK];
  __shared__ _Float16 Bs[TN * LDK];
  const int tid = threadIdx.x, lane = tid & 31, wave = tid >> 5;
  const int wm = wave >> 1, wn = wave & 1;
  const int mt = blockIdx.x, pt = blockIdx.y, n = blockIdx.z;
  const int r = tid >> 1, ko = (tid & 1) * 16;
  const _Float16* Arow = v + ((size_t)n * CC + mt * TM + r) * DPAD;
  const _Float16* Brow = probs + ((size_t)n * PPAD + pt * TN + r) * DPAD;
  v8f acc[2][4]; acc_zero(acc);

  uint4 a0, a1, b0, b1;
  g_ld32(Arow + ko, a0, a1);
  g_ld32(Brow + ko, b0, b1);
  for (int kb = 0; kb < DPAD / 32; ++kb) {
    __syncthreads();
    lds_st32(As, r, ko, a0, a1);
    lds_st32(Bs, r, ko, b0, b1);
    __syncthreads();
    if (kb + 1 < DPAD / 32) {
      g_ld32(Arow + (kb + 1) * 32 + ko, a0, a1);
      g_ld32(Brow + (kb + 1) * 32 + ko, b0, b1);
    }
    wmma_tile8(As, Bs, wm, wn, lane, acc);
  }
  const int mb = mt * TM + wm * 32, nb = pt * TN + wn * 64;
#pragma unroll
  for (int i = 0; i < 2; ++i)
#pragma unroll
    for (int j = 0; j < 4; ++j) {
      const int m = mb + i * 16 + ((lane & 16) ? 8 : 0);
      const int pr = nb + j * 16 + (lane & 15);
      st8h(a_t + ((size_t)n * PPAD + pr) * CC + m, acc[i][j]);  // pad rows -> zeros
    }
}

// ---------------- GEMM 5: gate preact = [Wx_g | Wa_g] * [im2col(xh) ; a] + b  (K=4864) ----------------

__global__ __launch_bounds__(256) void k_gate(const _Float16* __restrict__ Wgp,
                                              const _Float16* __restrict__ xh_t,
                                              const _Float16* __restrict__ a_t,
                                              const float* __restrict__ bias,
                                              float* __restrict__ pre) {
  __shared__ _Float16 As[TM * LDK];
  __shared__ _Float16 Bs[TN * LDK];
  const int tid = threadIdx.x, lane = tid & 31, wave = tid >> 5;
  const int wm = wave >> 1, wn = wave & 1;
  const int mt = blockIdx.x, pt = blockIdx.y, n = blockIdx.z;
  const int r = tid >> 1, ko = (tid & 1) * 16;
  const int p = pt * TN + r;
  const int y = p / WW, x = p % WW;
  const bool pv = (p < NPIX);
  const _Float16* Arow = Wgp + (size_t)(mt * TM + r) * KGATE;
  const _Float16* Xb = xh_t + (size_t)n * PPAD * TWOC;
  const _Float16* Ab = a_t + ((size_t)n * PPAD + p) * CC;
  v8f acc[2][4]; acc_zero(acc);

  uint4 a0, a1, b0, b1;
  auto loadB = [&](int kb, uint4& d0, uint4& d1) {
    if (kb < 144) {
      const int tap = kb >> 4;
      const int iy = y + tap / 3 - 1, ix = x + tap % 3 - 1;
      const bool ok = pv && iy >= 0 && iy < HH && ix >= 0 && ix < WW;
      const int prow = ok ? iy * WW + ix : 0;
      g_ld32(Xb + (size_t)prow * TWOC + (kb & 15) * 32 + ko, d0, d1);
      if (!ok) zero32(d0, d1);
    } else {
      g_ld32(Ab + (kb - 144) * 32 + ko, d0, d1);   // pad rows of a_t are zero
    }
  };
  g_ld32(Arow + ko, a0, a1);
  loadB(0, b0, b1);
  for (int kb = 0; kb < 152; ++kb) {
    __syncthreads();
    lds_st32(As, r, ko, a0, a1);
    lds_st32(Bs, r, ko, b0, b1);
    __syncthreads();
    if (kb + 1 < 152) {
      g_ld32(Arow + (kb + 1) * 32 + ko, a0, a1);
      loadB(kb + 1, b0, b1);
    }
    wmma_tile8(As, Bs, wm, wn, lane, acc);
  }
  const int mb = mt * TM + wm * 32, nb = pt * TN + wn * 64;
#pragma unroll
  for (int i = 0; i < 2; ++i)
#pragma unroll
    for (int j = 0; j < 4; ++j)
#pragma unroll
      for (int rr = 0; rr < 8; ++rr) {
        const int m = mb + i * 16 + rr + ((lane & 16) ? 8 : 0);
        const int pc = nb + j * 16 + (lane & 15);
        if (pc < NPIX)
          pre[((size_t)n * CC + m) * NPIX + pc] = acc[i][j][rr] + bias[m];
      }
}

// ---------------- fused LSTM pointwise ----------------

__global__ void k_lstm(const float* __restrict__ pi, const float* __restrict__ pf,
                       const float* __restrict__ pg, const float* __restrict__ po,
                       const float* __restrict__ c0, float* __restrict__ h) {
  const size_t total = (size_t)NB * CC * NPIX;
  size_t idx = (size_t)blockIdx.x * blockDim.x + threadIdx.x;
  if (idx >= total) return;
  const float ig = 1.f / (1.f + __expf(-pi[idx]));
  const float fg = 1.f / (1.f + __expf(-pf[idx]));
  const float gg = tanhf(pg[idx]);
  const float og = 1.f / (1.f + __expf(-po[idx]));
  const float c = fg * c0[idx] + ig * gg;
  h[idx] = og * tanhf(c);
}

// ---------------- Host orchestration ----------------

extern "C" void kernel_launch(void* const* d_in, const int* in_sizes, int n_in,
                              void* d_out, int out_size, void* d_ws, size_t ws_size,
                              hipStream_t stream) {
  (void)in_sizes; (void)n_in; (void)out_size; (void)ws_size;
  const float* x_in = (const float*)d_in[0];
  const float* h0   = (const float*)d_in[1];
  const float* c0   = (const float*)d_in[2];
  const float* tau  = (const float*)d_in[3];
  const float* Wx   = (const float*)d_in[4];
  const float* Wq   = (const float*)d_in[5];
  const float* Wk   = (const float*)d_in[6];
  const float* Wv   = (const float*)d_in[7];
  const float* Wa_g[4] = { (const float*)d_in[8],  (const float*)d_in[11],
                           (const float*)d_in[14], (const float*)d_in[17] };
  const float* Wx_g[4] = { (const float*)d_in[9],  (const float*)d_in[12],
                           (const float*)d_in[15], (const float*)d_in[18] };
  const float* b_g[4]  = { (const float*)d_in[10], (const float*)d_in[13],
                           (const float*)d_in[16], (const float*)d_in[19] };
  float* h_out = (float*)d_out;

  char* ws = (char*)d_ws;
  size_t off = 0;
  auto take = [&](size_t bytes) -> void* {
    void* p = ws + off;
    off = (off + bytes + 255) & ~(size_t)255;
    return p;
  };
  _Float16* xin_t = (_Float16*)take((size_t)NB * PPAD * CIN * 2);
  _Float16* xh_t  = (_Float16*)take((size_t)NB * PPAD * TWOC * 2);
  _Float16* Wxp   = (_Float16*)take((size_t)CC * CIN * 2);
  _Float16* Wqp   = (_Float16*)take((size_t)CC * KCONV * 2);
  _Float16* Wkp   = (_Float16*)take((size_t)CC * KCONV * 2);
  _Float16* Wvp   = (_Float16*)take((size_t)CC * KCONV * 2);
  _Float16* Wgp[4];
  for (int g = 0; g < 4; ++g) Wgp[g] = (_Float16*)take((size_t)CC * KGATE * 2);
  _Float16* q_t   = (_Float16*)take((size_t)NB * PPAD * CC * 2);
  _Float16* k_t   = (_Float16*)take((size_t)NB * DPAD * CC * 2);
  _Float16* vbuf  = (_Float16*)take((size_t)NB * CC * DPAD * 2);
  float*    scores= (float*)   take((size_t)NB * NPIX * DPAD * 4);
  _Float16* probs = (_Float16*)take((size_t)NB * PPAD * DPAD * 2);
  _Float16* a_t   = (_Float16*)take((size_t)NB * PPAD * CC * 2);
  float* pre = scores;                               // alias: lifetimes disjoint (85MB <= 106MB)
  const size_t pre_stride = (size_t)NB * CC * NPIX;

  const int MT      = CC / TM;                       // 2
  const int PT_SAME  = PPAD / TN;                    // 11
  const int PT_VALID = DPAD / TN;                    // 10
  const size_t elems = (size_t)NB * CC * NPIX;
  const int eblocks = (int)((elems + 255) / 256);

  // ---- prep: transposes, weight packs, probs pad fill ----
  k_tr<<<dim3(PPAD / 32, CIN / 32, NB), dim3(32, 8), 0, stream>>>(x_in, xin_t, CIN, CIN, 0);
  k_tr<<<dim3(PPAD / 32, CC / 32, NB), dim3(32, 8), 0, stream>>>(h0, xh_t, CC, TWOC, CC);
  k_pack_lin<<<(CC * CIN + 255) / 256, 256, 0, stream>>>(Wx, Wxp, CC * CIN);
  k_pack_conv<<<((CC * KCONV) + 255) / 256, 256, 0, stream>>>(Wq, nullptr, Wqp, KCONV);
  k_pack_conv<<<((CC * KCONV) + 255) / 256, 256, 0, stream>>>(Wk, nullptr, Wkp, KCONV);
  k_pack_conv<<<((CC * KCONV) + 255) / 256, 256, 0, stream>>>(Wv, nullptr, Wvp, KCONV);
  for (int g = 0; g < 4; ++g)
    k_pack_conv<<<((CC * KGATE) + 255) / 256, 256, 0, stream>>>(Wx_g[g], Wa_g[g], Wgp[g], KGATE);
  const size_t probs16 = (size_t)NB * PPAD * DPAD * 2 / 16;
  k_fill16<<<(int)((probs16 + 255) / 256), 256, 0, stream>>>((uint4*)probs, probs16);

  // ---- xh_t channels 0..255 = Wx * x_in ----
  k_proj<<<dim3(MT, PT_SAME, NB), 256, 0, stream>>>(Wxp, xin_t, xh_t);

  // ---- q (SAME, transposed), k (VALID, transposed), v (VALID, normal) ----
  k_conv3<<<dim3(MT, PT_SAME,  NB), 256, 0, stream>>>(Wqp, xh_t, q_t,  WW,  NPIX, 1, 1, PPAD);
  k_conv3<<<dim3(MT, PT_VALID, NB), 256, 0, stream>>>(Wkp, xh_t, k_t,  DHV, ND,   0, 1, DPAD);
  k_conv3<<<dim3(MT, PT_VALID, NB), 256, 0, stream>>>(Wvp, xh_t, vbuf, DHV, ND,   0, 0, 0);

  // ---- attention ----
  k_scores<<<dim3(PPAD / TM, PT_VALID, NB), 256, 0, stream>>>(q_t, k_t, tau, scores);
  k_softmax<<<NB * NPIX, 256, 0, stream>>>(scores, probs);
  k_apply<<<dim3(MT, PT_SAME, NB), 256, 0, stream>>>(vbuf, probs, a_t);

  // ---- gates (scores buffer reused as pre-activations) ----
  for (int g = 0; g < 4; ++g)
    k_gate<<<dim3(MT, PT_SAME, NB), 256, 0, stream>>>(
        Wgp[g], xh_t, a_t, b_g[g], pre + (size_t)g * pre_stride);

  // ---- fused LSTM pointwise -> h ----
  k_lstm<<<eblocks, 256, 0, stream>>>(pre + 0 * pre_stride, pre + 1 * pre_stride,
                                      pre + 2 * pre_stride, pre + 3 * pre_stride,
                                      c0, h_out);
}
